// RayTracing_85358180040810
// MI455X (gfx1250) — matrix operand order repaired and checked by reading, not compiled, not run on hardware
//
#include <hip/hip_runtime.h>
#include <hip/hip_bf16.h>
#include <math.h>

#define HID     256
#define NB      2
#define NN      2048
#define NRAYS   (NB*NN)          // 4096
#define NSTEPS  100
#define NSAMP   (NRAYS*NSTEPS)   // 409600
#define TILE    64               // points per block in the MLP kernel
#define SDF_THRv 5e-5f
#define LS_STEP  0.5f
#define NSECANT  8
#define NTRACE   10

typedef __attribute__((ext_vector_type(16))) _Float16 v16h;
typedef __attribute__((ext_vector_type(8)))  _Float16 v8h;
typedef __attribute__((ext_vector_type(8)))  float    v8f;

__device__ __forceinline__ float softplus_f(float x) {
    // stable softplus on the HW transcendental pipe: max(x,0) + log(1+exp(-|x|))
    return fmaxf(x, 0.f) + __logf(1.0f + __expf(-fabsf(x)));
}

// Load the 8 B fragments (one 32x16 K-chunk each) of a 16-column tile.
__device__ __forceinline__ void load_bfrag(const _Float16* __restrict__ Bt, v16h (&buf)[8]) {
#pragma unroll
    for (int kk = 0; kk < 8; ++kk) {
        const _Float16* p = Bt + kk * 32;
        v8h blo = *(const v8h*)p;
        v8h bhi = *(const v8h*)(p + 8);
        buf[kk] = __builtin_shufflevector(blo, bhi, 0,1,2,3,4,5,6,7,8,9,10,11,12,13,14,15);
    }
}

// One 16x16 output tile: 8-step WMMA chain + fused bias/softplus/w3 epilogue.
__device__ __forceinline__ void tile_mma(const v16h (&afr)[8], const v16h (&b)[8],
                                         int n, const float* __restrict__ w3,
                                         const float* __restrict__ b2, float (&partial)[8]) {
    v8f acc = {0.f, 0.f, 0.f, 0.f, 0.f, 0.f, 0.f, 0.f};
#pragma unroll
    for (int kk = 0; kk < 8; ++kk)
        acc = __builtin_amdgcn_wmma_f32_16x16x32_f16(false, afr[kk], false, b[kk],
                                                     (short)0, acc, false, false);
    const float w3n = w3[n], b2n = b2[n];
#pragma unroll
    for (int i = 0; i < 8; ++i)
        partial[i] += softplus_f(acc[i] + b2n) * w3n;
}

// ---------------------------------------------------------------------------
// Batched SDF MLP: out[i] = norm(p_i) - 0.9 + 0.1*(softplus(softplus(p W1+b1) W2+b2)@w3 + b3)
// Block = 128 threads (4 waves). 64 points per block. W2 GEMM on WMMA f16->f32.
// B fragments ping-pong between two register buffers (no copies, no WMMA hazards);
// the WMMA chain only waits on loads issued a full tile earlier.
// ---------------------------------------------------------------------------
__global__ __launch_bounds__(128)
void sdf_mlp_kernel(const float* __restrict__ pts, float* __restrict__ out,
                    const float* __restrict__ W1, const float* __restrict__ b1,
                    const _Float16* __restrict__ W2T,  // [n][k] f16, k contiguous
                    const float* __restrict__ b2,
                    const float* __restrict__ w3, const float* __restrict__ b3p)
{
    __shared__ __align__(16) float ptile[TILE * 3];
    __shared__ float pnorm[TILE];
    __shared__ float W1s[3 * HID];
    __shared__ float b1s[HID];
    __shared__ __align__(16) _Float16 h1s[TILE][HID + 8]; // +8 halves pad -> no bank conflicts

    const int t    = threadIdx.x;
    const int base = blockIdx.x * TILE;

    // CDNA5 async copy: stage the 64x3 f32 point tile (768B) straight into LDS,
    // tracked by ASYNCcnt (no VGPR round-trip).
    if (t < 48) {
        const char* gsrc = (const char*)(pts + (size_t)base * 3) + t * 16;
        unsigned    ldst = (unsigned)(uintptr_t)((const char*)ptile + t * 16);
        asm volatile("global_load_async_to_lds_b128 %0, %1, off"
                     :: "v"(ldst), "v"(gsrc) : "memory");
    }
    for (int i = t; i < 3 * HID; i += 128) W1s[i] = W1[i];
    for (int i = t; i < HID;     i += 128) b1s[i] = b1[i];
    asm volatile("s_wait_asynccnt 0x0" ::: "memory");
    __syncthreads();

    if (t < TILE) {
        float p0 = ptile[t * 3 + 0], p1 = ptile[t * 3 + 1], p2 = ptile[t * 3 + 2];
        pnorm[t] = sqrtf(p0 * p0 + p1 * p1 + p2 * p2 + 1e-12f);
    }
    // Layer 1 (K=3) on VALU, f16 result staged to LDS
    for (int idx = t; idx < TILE * HID; idx += 128) {
        int r = idx >> 8, c = idx & (HID - 1);
        float x = ptile[r * 3 + 0] * W1s[c]
                + ptile[r * 3 + 1] * W1s[HID + c]
                + ptile[r * 3 + 2] * W1s[2 * HID + c] + b1s[c];
        h1s[r][c] = (_Float16)softplus_f(x);
    }
    __syncthreads();

    // Layer 2: each wave owns 16 rows; 16 col-tiles x 8 K-steps of WMMA
    const int wave  = t >> 5;
    const int lane  = t & 31;
    const int mBase = wave * 16;
    const int row   = mBase + (lane & 15);
    const int hi    = lane >> 4;           // K half-select: this code's contiguous k = 16*hi + s order (A and B alike)
    const int nlane = lane & 15;
    const float b3  = b3p[0];

    // Hoist all 8 A fragments (full K=256) for this wave's 16 rows into registers
    v16h afr[8];
#pragma unroll
    for (int kk = 0; kk < 8; ++kk) {
        const int k0 = kk * 32;
        v8h alo = *(const v8h*)&h1s[row][k0 + hi * 16];
        v8h ahi = *(const v8h*)&h1s[row][k0 + hi * 16 + 8];
        afr[kk] = __builtin_shufflevector(alo, ahi, 0,1,2,3,4,5,6,7,8,9,10,11,12,13,14,15);
    }

    float partial[8];
#pragma unroll
    for (int i = 0; i < 8; ++i) partial[i] = 0.f;

    const _Float16* __restrict__ Bbase = W2T + (size_t)nlane * HID + hi * 16;

    v16h bufA[8], bufB[8];
    load_bfrag(Bbase, bufA);                                   // tile n0 = 0

    for (int n0 = 0; n0 < HID; n0 += 32) {
        // fill B while A-buffer's tile computes, then swap roles (no reg copies)
        load_bfrag(Bbase + (size_t)(n0 + 16) * HID, bufB);
        __builtin_prefetch(Bbase + (size_t)((n0 + 64) & (HID - 1)) * HID, 0, 3);
        tile_mma(afr, bufA, n0 + nlane, w3, b2, partial);

        load_bfrag(Bbase + (size_t)((n0 + 32) & (HID - 1)) * HID, bufA);
        tile_mma(afr, bufB, n0 + 16 + nlane, w3, b2, partial);
    }

    // Reduce over the 16 lanes of each half (wave32 shuffles)
#pragma unroll
    for (int m = 8; m >= 1; m >>= 1)
#pragma unroll
        for (int i = 0; i < 8; ++i)
            partial[i] += __shfl_xor(partial[i], m, 32);

    if (nlane == 0) {
#pragma unroll
        for (int i = 0; i < 8; ++i) {
            int r = mBase + hi * 8 + i;   // C layout: lanes<16 rows 0-7, lanes>=16 rows 8-15
            out[base + r] = pnorm[r] - 0.9f + 0.1f * (partial[i] + b3);
        }
    }
}

// ---------------------------------------------------------------------------
// Small orchestration kernels
// ---------------------------------------------------------------------------
__global__ void k_convert_w2(const float* __restrict__ W2, _Float16* __restrict__ W2T)
{
    int k = blockIdx.x, n = threadIdx.x;                 // W2 is [K][N]
    W2T[(size_t)n * HID + k] = (_Float16)W2[(size_t)k * HID + n];
}

__global__ void k_init(const float* __restrict__ cam, const float* __restrict__ dirs,
                       float* curr_pts, float* acc_start, float* acc_end,
                       int* unfinished, float* pts)
{
    int i = blockIdx.x * blockDim.x + threadIdx.x;
    if (i >= NRAYS) return;
    int b = i >> 11;
    float cx = cam[b*3], cy = cam[b*3+1], cz = cam[b*3+2];
    float dx = dirs[i*3], dy = dirs[i*3+1], dz = dirs[i*3+2];
    float rcd = dx*cx + dy*cy + dz*cz;
    float under = rcd*rcd - (cx*cx + cy*cy + cz*cz - 1.0f);
    int m = under > 0.f;
    float s = sqrtf(m ? under : 1.f);
    float d0 = m ? fmaxf(-s - rcd, 0.f) : 0.f;
    float d1 = m ? fmaxf( s - rcd, 0.f) : 0.f;
    float px = m ? cx + d0*dx : 0.f;
    float py = m ? cy + d0*dy : 0.f;
    float pz = m ? cz + d0*dz : 0.f;
    curr_pts[i*3+0] = px; curr_pts[i*3+1] = py; curr_pts[i*3+2] = pz;
    pts[i*3+0] = px; pts[i*3+1] = py; pts[i*3+2] = pz;
    acc_start[i] = d0; acc_end[i] = d1; unfinished[i] = m;
}

__global__ void k_mask_next(const float* __restrict__ sdfv, const int* __restrict__ unf,
                            float* next_sdf)
{
    int i = blockIdx.x * blockDim.x + threadIdx.x;
    if (i >= NRAYS) return;
    next_sdf[i] = unf[i] ? sdfv[i] : 0.f;
}

__global__ void k_trace_pre(const float* __restrict__ next_sdf, int* unfinished,
                            float* curr_sdf, int* flagArr)
{
    __shared__ int any_s;
    int i = blockIdx.x * blockDim.x + threadIdx.x;
    if (threadIdx.x == 0) any_s = 0;
    __syncthreads();
    int unf = 0;
    if (i < NRAYS) {
        unf = unfinished[i];
        float cs = unf ? next_sdf[i] : 0.f;
        cs = (cs <= SDF_THRv) ? 0.f : cs;
        unf = unf && (cs > SDF_THRv);
        unfinished[i] = unf;
        curr_sdf[i] = cs;
    }
    if (unf) atomicOr(&any_s, 1);
    __syncthreads();
    if (threadIdx.x == 0) flagArr[blockIdx.x] = any_s;
}

__global__ void k_trace_step1(const float* __restrict__ cam, const float* __restrict__ dirs,
                              const float* __restrict__ acc_start, const float* __restrict__ curr_sdf,
                              float* acc_new, float* pts_new)
{
    int i = blockIdx.x * blockDim.x + threadIdx.x;
    if (i >= NRAYS) return;
    int b = i >> 11;
    float an = acc_start[i] + curr_sdf[i];
    acc_new[i] = an;
    pts_new[i*3+0] = cam[b*3+0] + an * dirs[i*3+0];
    pts_new[i*3+1] = cam[b*3+1] + an * dirs[i*3+1];
    pts_new[i*3+2] = cam[b*3+2] + an * dirs[i*3+2];
}

__global__ void k_trace_ls(const float* __restrict__ cam, const float* __restrict__ dirs,
                           const float* __restrict__ sdfv, const int* __restrict__ unf,
                           const float* __restrict__ curr_sdf,
                           float* acc_new, float* nsdf_arr, int* not_proj, float* pts_new)
{
    int i = blockIdx.x * blockDim.x + threadIdx.x;
    if (i >= NRAYS) return;
    float ns = unf[i] ? sdfv[i] : 0.f;
    int np = ns < 0.f;
    nsdf_arr[i] = ns; not_proj[i] = np;
    if (np) {
        float an = acc_new[i] - (1.0f - LS_STEP) * curr_sdf[i];
        acc_new[i] = an;
        int b = i >> 11;
        pts_new[i*3+0] = cam[b*3+0] + an * dirs[i*3+0];
        pts_new[i*3+1] = cam[b*3+1] + an * dirs[i*3+1];
        pts_new[i*3+2] = cam[b*3+2] + an * dirs[i*3+2];
    }
}

__global__ void k_trace_post(const int* __restrict__ flagArr, const float* __restrict__ sdfv,
                             const float* __restrict__ nsdf_arr, const int* __restrict__ not_proj,
                             const float* __restrict__ acc_new, const float* __restrict__ pts_new,
                             const float* __restrict__ acc_end,
                             float* acc_start, float* curr_pts, float* next_sdf, int* unfinished)
{
    int i = blockIdx.x * blockDim.x + threadIdx.x;
    if (i >= NRAYS) return;
    int active = 0;
#pragma unroll
    for (int b = 0; b < 16; ++b) active |= flagArr[b];
    if (!active) return;
    next_sdf[i]  = not_proj[i] ? sdfv[i] : nsdf_arr[i];
    float an = acc_new[i];
    acc_start[i] = an;
    curr_pts[i*3+0] = pts_new[i*3+0];
    curr_pts[i*3+1] = pts_new[i*3+1];
    curr_pts[i*3+2] = pts_new[i*3+2];
    unfinished[i] = unfinished[i] && (an < acc_end[i]);
}

__global__ void k_trace_final(const float* __restrict__ next_sdf, int* unfinished)
{
    int i = blockIdx.x * blockDim.x + threadIdx.x;
    if (i >= NRAYS) return;
    int unf = unfinished[i];
    float cs = unf ? next_sdf[i] : 0.f;
    cs = (cs <= SDF_THRv) ? 0.f : cs;
    unfinished[i] = unf && (cs > SDF_THRv);
}

__global__ void k_gen_samples(const float* __restrict__ cam, const float* __restrict__ dirs,
                              const int* __restrict__ smask,
                              const float* __restrict__ acc_start, const float* __restrict__ acc_end,
                              float* pts, float* pint_buf)
{
    int i = blockIdx.x * blockDim.x + threadIdx.x;
    if (i >= NSAMP) return;
    int ray = i / NSTEPS, s = i - ray * NSTEPS;
    int m = smask[ray];
    float mm0 = m ? acc_start[ray] : 0.f;
    float mm1 = m ? acc_end[ray]   : 0.f;
    float tt = (float)s * (1.0f / (float)(NSTEPS - 1));
    float pint = mm0 + tt * (mm1 - mm0);
    int b = ray >> 11;
    pts[i*3+0] = cam[b*3+0] + pint * dirs[ray*3+0];
    pts[i*3+1] = cam[b*3+1] + pint * dirs[ray*3+1];
    pts[i*3+2] = cam[b*3+2] + pint * dirs[ray*3+2];
    pint_buf[i] = pint;
}

__global__ void k_sampler_select(const int* __restrict__ omask, const int* __restrict__ smask,
                                 const float* __restrict__ sdfv, const float* __restrict__ pint_buf,
                                 const float* __restrict__ pts,
                                 float* samp_pts, float* samp_dists, int* samp_net, int* secant,
                                 float* zl, float* zh, float* sl, float* sh)
{
    int r = blockIdx.x * blockDim.x + threadIdx.x;
    if (r >= NRAYS) return;
    int base = r * NSTEPS;
    float best = 1e30f, bestsdf = 1e30f;
    int ind = 0, out_ind = 0;
    for (int s = 0; s < NSTEPS; ++s) {
        float v = sdfv[base + s];
        float sgn = (float)((v > 0.f) - (v < 0.f));
        float tmp = sgn * (float)(NSTEPS - s);
        if (tmp < best) { best = tmp; ind = s; }
        if (v < bestsdf) { bestsdf = v; out_ind = s; }
    }
    float sel = sdfv[base + ind];
    int net_surface  = sel < 0.f;
    int true_surface = omask[r] != 0;
    int p_out = !(true_surface && net_surface);
    int ind2 = p_out ? out_ind : ind;
    int m = smask[r];
    samp_pts[r*3+0] = m ? pts[(base + ind2)*3+0] : 0.f;
    samp_pts[r*3+1] = m ? pts[(base + ind2)*3+1] : 0.f;
    samp_pts[r*3+2] = m ? pts[(base + ind2)*3+2] : 0.f;
    samp_dists[r] = m ? pint_buf[base + ind2] : 0.f;
    samp_net[r] = m && net_surface;
    secant[r] = net_surface && true_surface && m;
    int indm1 = (ind + NSTEPS - 1) % NSTEPS;     // torch-style -1 wrap
    zh[r] = pint_buf[base + ind];    sh[r] = sdfv[base + ind];
    zl[r] = pint_buf[base + indm1];  sl[r] = sdfv[base + indm1];
}

__device__ __forceinline__ float zpred_f(float zl, float zh, float sl, float sh) {
    float denom = sh - sl;
    denom = (denom == 0.f) ? 1.f : denom;
    return -sl * (zh - zl) / denom + zl;
}

__global__ void k_secant_gen(const float* __restrict__ cam, const float* __restrict__ dirs,
                             const float* __restrict__ zl, const float* __restrict__ zh,
                             const float* __restrict__ sl, const float* __restrict__ sh,
                             float* zpred, float* pts)
{
    int i = blockIdx.x * blockDim.x + threadIdx.x;
    if (i >= NRAYS) return;
    float zp = zpred_f(zl[i], zh[i], sl[i], sh[i]);
    zpred[i] = zp;
    int b = i >> 11;
    pts[i*3+0] = cam[b*3+0] + zp * dirs[i*3+0];
    pts[i*3+1] = cam[b*3+1] + zp * dirs[i*3+1];
    pts[i*3+2] = cam[b*3+2] + zp * dirs[i*3+2];
}

__global__ void k_secant_upd(const float* __restrict__ sdfv, const float* __restrict__ zpred,
                             float* zl, float* zh, float* sl, float* sh)
{
    int i = blockIdx.x * blockDim.x + threadIdx.x;
    if (i >= NRAYS) return;
    float mid = sdfv[i], zp = zpred[i];
    if (mid > 0.f) { zl[i] = zp; sl[i] = mid; }
    if (mid < 0.f) { zh[i] = zp; sh[i] = mid; }
}

__global__ void k_finalize(const float* __restrict__ cam, const float* __restrict__ dirs,
                           const int* __restrict__ secant,
                           const float* __restrict__ zl, const float* __restrict__ zh,
                           const float* __restrict__ sl, const float* __restrict__ sh,
                           const float* __restrict__ samp_pts, const float* __restrict__ samp_dists,
                           const int* __restrict__ samp_net, const int* __restrict__ smask,
                           const float* __restrict__ acc_start, const float* __restrict__ acc_end,
                           const float* __restrict__ curr_pts, float* __restrict__ out)
{
    int i = blockIdx.x * blockDim.x + threadIdx.x;
    if (i >= NRAYS) return;
    float zp = zpred_f(zl[i], zh[i], sl[i], sh[i]);
    int b = i >> 11;
    float sx = cam[b*3+0] + zp * dirs[i*3+0];
    float sy = cam[b*3+1] + zp * dirs[i*3+1];
    float sz = cam[b*3+2] + zp * dirs[i*3+2];
    int sec = secant[i];
    float spx = sec ? sx : samp_pts[i*3+0];
    float spy = sec ? sy : samp_pts[i*3+1];
    float spz = sec ? sz : samp_pts[i*3+2];
    float sdist = sec ? zp : samp_dists[i];
    int m = smask[i];
    out[i*3+0] = m ? spx : curr_pts[i*3+0];
    out[i*3+1] = m ? spy : curr_pts[i*3+1];
    out[i*3+2] = m ? spz : curr_pts[i*3+2];
    int netm = m ? samp_net[i] : (acc_start[i] < acc_end[i]);
    out[3*NRAYS + i] = netm ? 1.f : 0.f;
    out[4*NRAYS + i] = m ? sdist : acc_start[i];
}

// ---------------------------------------------------------------------------
extern "C" void kernel_launch(void* const* d_in, const int* in_sizes, int n_in,
                              void* d_out, int out_size, void* d_ws, size_t ws_size,
                              hipStream_t stream)
{
    const float* cam   = (const float*)d_in[0];
    const int*   omask = (const int*)  d_in[1];
    const float* dirs  = (const float*)d_in[2];
    const float* W1    = (const float*)d_in[3];
    const float* b1    = (const float*)d_in[4];
    const float* W2    = (const float*)d_in[5];
    const float* b2    = (const float*)d_in[6];
    const float* w3    = (const float*)d_in[7];
    const float* b3    = (const float*)d_in[8];
    float* out = (float*)d_out;

    char* ws = (char*)d_ws;
    size_t off = 0;
    auto take = [&](size_t bytes) -> char* {
        char* p = ws + off;
        off = (off + bytes + 255) & ~(size_t)255;
        return p;
    };
    _Float16* W2T   = (_Float16*)take((size_t)HID * HID * 2);
    float* pts_big  = (float*)take((size_t)NSAMP * 3 * 4);
    float* sdf_buf  = (float*)take((size_t)NSAMP * 4);
    float* pint_buf = (float*)take((size_t)NSAMP * 4);
    float* curr_pts = (float*)take(NRAYS * 3 * 4);
    float* pts_new  = (float*)take(NRAYS * 3 * 4);
    float* samp_pts = (float*)take(NRAYS * 3 * 4);
    float* acc_start= (float*)take(NRAYS * 4);
    float* acc_end  = (float*)take(NRAYS * 4);
    float* next_sdf = (float*)take(NRAYS * 4);
    float* curr_sdf = (float*)take(NRAYS * 4);
    float* acc_new  = (float*)take(NRAYS * 4);
    float* nsdf_arr = (float*)take(NRAYS * 4);
    float* samp_dists = (float*)take(NRAYS * 4);
    float* zl = (float*)take(NRAYS * 4);
    float* zh = (float*)take(NRAYS * 4);
    float* sl = (float*)take(NRAYS * 4);
    float* sh = (float*)take(NRAYS * 4);
    float* zpred = (float*)take(NRAYS * 4);
    int* unfinished = (int*)take(NRAYS * 4);
    int* not_proj   = (int*)take(NRAYS * 4);
    int* samp_net   = (int*)take(NRAYS * 4);
    int* secant     = (int*)take(NRAYS * 4);
    int* flagArr    = (int*)take(16 * 4);

    const dim3 rblk(256), rgrd(NRAYS / 256);             // 16 blocks of 256

    k_convert_w2<<<HID, HID, 0, stream>>>(W2, W2T);
    k_init<<<rgrd, rblk, 0, stream>>>(cam, dirs, curr_pts, acc_start, acc_end, unfinished, pts_new);
    sdf_mlp_kernel<<<NRAYS / TILE, 128, 0, stream>>>(pts_new, sdf_buf, W1, b1, W2T, b2, w3, b3);
    k_mask_next<<<rgrd, rblk, 0, stream>>>(sdf_buf, unfinished, next_sdf);

    for (int it = 0; it < NTRACE; ++it) {
        k_trace_pre<<<rgrd, rblk, 0, stream>>>(next_sdf, unfinished, curr_sdf, flagArr);
        k_trace_step1<<<rgrd, rblk, 0, stream>>>(cam, dirs, acc_start, curr_sdf, acc_new, pts_new);
        sdf_mlp_kernel<<<NRAYS / TILE, 128, 0, stream>>>(pts_new, sdf_buf, W1, b1, W2T, b2, w3, b3);
        k_trace_ls<<<rgrd, rblk, 0, stream>>>(cam, dirs, sdf_buf, unfinished, curr_sdf,
                                              acc_new, nsdf_arr, not_proj, pts_new);
        sdf_mlp_kernel<<<NRAYS / TILE, 128, 0, stream>>>(pts_new, sdf_buf, W1, b1, W2T, b2, w3, b3);
        k_trace_post<<<rgrd, rblk, 0, stream>>>(flagArr, sdf_buf, nsdf_arr, not_proj, acc_new,
                                                pts_new, acc_end, acc_start, curr_pts,
                                                next_sdf, unfinished);
    }
    k_trace_final<<<rgrd, rblk, 0, stream>>>(next_sdf, unfinished);

    // Ray sampler: 409,600 SDF evals -> the dominant WMMA GEMM
    k_gen_samples<<<NSAMP / 256, 256, 0, stream>>>(cam, dirs, unfinished, acc_start, acc_end,
                                                   pts_big, pint_buf);
    sdf_mlp_kernel<<<NSAMP / TILE, 128, 0, stream>>>(pts_big, sdf_buf, W1, b1, W2T, b2, w3, b3);
    k_sampler_select<<<rgrd, rblk, 0, stream>>>(omask, unfinished, sdf_buf, pint_buf, pts_big,
                                                samp_pts, samp_dists, samp_net, secant,
                                                zl, zh, sl, sh);

    for (int it = 0; it < NSECANT; ++it) {
        k_secant_gen<<<rgrd, rblk, 0, stream>>>(cam, dirs, zl, zh, sl, sh, zpred, pts_new);
        sdf_mlp_kernel<<<NRAYS / TILE, 128, 0, stream>>>(pts_new, sdf_buf, W1, b1, W2T, b2, w3, b3);
        k_secant_upd<<<rgrd, rblk, 0, stream>>>(sdf_buf, zpred, zl, zh, sl, sh);
    }

    k_finalize<<<rgrd, rblk, 0, stream>>>(cam, dirs, secant, zl, zh, sl, sh,
                                          samp_pts, samp_dists, samp_net, unfinished,
                                          acc_start, acc_end, curr_pts, out);
    (void)in_sizes; (void)n_in; (void)out_size; (void)ws_size;
}
